// ADAP_2894807957856
// MI455X (gfx1250) — matrix operand order repaired and checked by reading, not compiled, not run on hardware
//
#include <hip/hip_runtime.h>

typedef __attribute__((ext_vector_type(16))) __bf16 v16bf;
typedef __attribute__((ext_vector_type(8)))  float  v8f;

#define HW   65536     // 256*256
#define NCH  18        // G*K*K
#define CSTR 72        // channel stride (halves) in raw tile: 144 B, 16B-aligned

__device__ __forceinline__ unsigned short f2bf(float x) {
  unsigned u = __float_as_uint(x);
  u += 0x7fffu + ((u >> 16) & 1u);          // round-to-nearest-even
  return (unsigned short)(u >> 16);
}
__device__ __forceinline__ int refl(int x) {      // 'reflect' pad, H=W=256, pad=1
  return x < 0 ? -x : (x > 255 ? 510 - x : x);
}

// ---------------------------------------------------------------------------
// Kernel 0: prearrange weights into WMMA A-fragment order (bf16) + zero stats.
// A 16x32 bf16 layout (contiguous, matches B): lane<16 -> M=lane, K=e (0..15);
// lane>=16 -> M=lane-16, K=16+e.
// K ordering tap-major: k = tap*64 + c  (tap=3*i+j) so each aligned 16-run of
// K is one tap with 16 consecutive channels.
// ---------------------------------------------------------------------------
__global__ __launch_bounds__(256) void adap_prep(const float* __restrict__ w,
                                                 unsigned short* __restrict__ wfrag,
                                                 float* __restrict__ stats) {
  int f = blockIdx.x * 256 + threadIdx.x;
  if (f < 36) stats[f] = 0.0f;
  if (f >= 2 * 18 * 32 * 16) return;
  int mtile = f / 9216;
  int rem   = f - mtile * 9216;
  int ks    = rem >> 9;          // /512
  int rem2  = rem & 511;
  int lane  = rem2 >> 4;
  int e     = rem2 & 15;
  int laneHi = lane >> 4;
  int m      = mtile * 16 + (lane & 15);
  int kl     = (laneHi << 4) + e;
  int k   = ks * 32 + kl;
  int tap = k >> 6;              // k = tap*64 + c
  int c   = k & 63;
  float val = (m < NCH) ? w[(m * 64 + c) * 9 + tap] : 0.0f;
  wfrag[f] = f2bf(val);
}

// ---------------------------------------------------------------------------
// Kernel 1: implicit-GEMM conv via v_wmma_f32_16x16x32_bf16.
// Block = (n, h, 64-pixel row segment). 4 waves x 16 pixels each.
// A fragments: async-copied global->LDS (ASYNCcnt). B fragments: raw tile in
// [row][col][c] order so one fragment = 2 x ds_load_b128.
// ---------------------------------------------------------------------------
__global__ __launch_bounds__(128) void adap_conv_wmma(const float* __restrict__ y,
                                                      const unsigned short* __restrict__ wfrag_g,
                                                      float* __restrict__ sigma,
                                                      float* __restrict__ stats) {
  __shared__ __align__(16) unsigned short raw2[3 * 66 * CSTR]; // 28512 B
  __shared__ uint4 wlds[2304];                                 // 36864 B
  __shared__ float ssum[NCH], ssq[NCH];

  int bid = blockIdx.x;                // 4096 = n(4) * h(256) * wt(4)
  int wt  = bid & 3;
  int h   = (bid >> 2) & 255;
  int n   = bid >> 10;
  int w0  = wt << 6;

  if (threadIdx.x < NCH) { ssum[threadIdx.x] = 0.f; ssq[threadIdx.x] = 0.f; }

  // A fragments: async global->LDS b128 copies (no VGPR round trip)
  const uint4* wg = (const uint4*)wfrag_g;
  for (int t = threadIdx.x; t < 2304; t += 128) {
    unsigned lds_off = (unsigned)(uintptr_t)(&wlds[t]);
    unsigned long long ga = (unsigned long long)(uintptr_t)(wg + t);
    asm volatile("global_load_async_to_lds_b128 %0, %1, off"
                 :: "v"(lds_off), "v"(ga) : "memory");
  }

  // stage reflect-padded raw tile: rows(h-1..h+1) x cols(w0-1..w0+64) x 64ch,
  // read with col-fastest (coalesced), store c-fastest (for b128 B fragments)
  const float* yn = y + ((size_t)n << 22);
  for (int t = threadIdx.x; t < 64 * 3 * 66; t += 128) {
    int c   = t / 198;
    int r2  = t - c * 198;
    int row = r2 / 66;
    int col = r2 - row * 66;
    int gr = refl(h - 1 + row);
    int gc = refl(w0 - 1 + col);
    raw2[(row * 66 + col) * CSTR + c] = f2bf(yn[(c << 16) + (gr << 8) + gc]);
  }
  asm volatile("s_wait_asynccnt 0x0" ::: "memory");
  __syncthreads();

  int lane     = threadIdx.x & 31;
  int waveId   = threadIdx.x >> 5;
  int pix      = lane & 15;
  int hiH      = lane >> 4;
  int wavebase = waveId << 4;

  v8f acc0 = {};
  v8f acc1 = {};

  for (int ks = 0; ks < 18; ++ks) {
    // A fragments from LDS (two b128 reads each)
    union { uint4 q[2]; v16bf v; } A0, A1;
    int f0 = (ks * 32 + lane) * 2;
    int f1 = ((18 + ks) * 32 + lane) * 2;
    A0.q[0] = wlds[f0]; A0.q[1] = wlds[f0 + 1];
    A1.q[0] = wlds[f1]; A1.q[1] = wlds[f1 + 1];

    // B fragment: K = ks*32 + (hiH?16:0) + e -> one tap, channels c0..c0+15
    // contiguous in LDS: exactly two b128 loads.
    int kb  = ks * 32 + (hiH << 4);
    int tap = kb >> 6;
    int c0  = kb & 63;
    int col = wavebase + pix + (tap % 3);
    int base = ((tap / 3) * 66 + col) * CSTR + c0;   // 16B-aligned
    union { uint4 q[2]; v16bf v; } B;
    const uint4* bp = (const uint4*)&raw2[base];
    B.q[0] = bp[0];
    B.q[1] = bp[1];

    acc0 = __builtin_amdgcn_wmma_f32_16x16x32_bf16(false, A0.v, false, B.v,
                                                   (short)0, acc0, false, false);
    acc1 = __builtin_amdgcn_wmma_f32_16x16x32_bf16(false, A1.v, false, B.v,
                                                   (short)0, acc1, false, false);
  }

  // Epilogue: C/D layout -> lane<16: M=r, lane>=16: M=8+r, N=lane&15.
  int colg = w0 + wavebase + pix;
  float* sp = sigma + (n * NCH << 16) + (h << 8) + colg;
#pragma unroll
  for (int r = 0; r < 8; ++r) {
    int m0 = r + (hiH << 3);
    float v0 = acc0[r];
    sp[m0 << 16] = v0;
    atomicAdd(&ssum[m0], v0);
    atomicAdd(&ssq[m0],  v0 * v0);
    int m1 = 16 + r + (hiH << 3);
    if (m1 < NCH) {
      float v1 = acc1[r];
      sp[m1 << 16] = v1;
      atomicAdd(&ssum[m1], v1);
      atomicAdd(&ssq[m1],  v1 * v1);
    }
  }
  __syncthreads();
  if (threadIdx.x < NCH) {
    atomicAdd(&stats[threadIdx.x],       ssum[threadIdx.x]);
    atomicAdd(&stats[NCH + threadIdx.x], ssq[threadIdx.x]);
  }
}

// ---------------------------------------------------------------------------
// Kernel 2: BN stats -> fused scale/shift
// ---------------------------------------------------------------------------
__global__ void adap_finalize(const float* __restrict__ stats,
                              const float* __restrict__ gamma,
                              const float* __restrict__ beta,
                              float* __restrict__ scsh) {
  int t = threadIdx.x;
  if (t < NCH) {
    const float inv_cnt = 1.0f / (4.0f * HW);
    float mean = stats[t] * inv_cnt;
    float var  = stats[NCH + t] * inv_cnt - mean * mean;
    float a = gamma[t] * rsqrtf(var + 1e-5f);
    scsh[t]       = a;
    scsh[NCH + t] = beta[t] - mean * a;
  }
}

// ---------------------------------------------------------------------------
// Kernel 3: per-pixel softmax over 18 channels (-> v_map) + 9-tap apply (-> out)
// Block = one (n, h) row of 256 pixels. y rows are L2-resident.
// ---------------------------------------------------------------------------
__global__ __launch_bounds__(256) void adap_apply(const float* __restrict__ y,
                                                  const float* __restrict__ sigma,
                                                  const float* __restrict__ scsh,
                                                  float* __restrict__ out,
                                                  float* __restrict__ vmap) {
  __shared__ float sc[36];
  if (threadIdx.x < 36) sc[threadIdx.x] = scsh[threadIdx.x];
  __syncthreads();

  int h  = blockIdx.x & 255;
  int n  = blockIdx.x >> 8;
  int wc = threadIdx.x;
  int q  = (h << 8) + wc;

  const float* sp = sigma + (n * NCH << 16) + q;
  float v[NCH];
  float mx = -3.0e38f;
#pragma unroll
  for (int m = 0; m < NCH; ++m) {
    float s = sp[m << 16] * sc[m] + sc[NCH + m];
    v[m] = s;
    mx = fmaxf(mx, s);
  }
  float sum = 0.0f;
#pragma unroll
  for (int m = 0; m < NCH; ++m) { v[m] = __expf(v[m] - mx); sum += v[m]; }
  float inv = 1.0f / sum;
  float* vp = vmap + (n * NCH << 16) + q;
#pragma unroll
  for (int m = 0; m < NCH; ++m) { v[m] *= inv; vp[m << 16] = v[m]; }

  int hm = (h == 0)   ? 1   : h - 1;
  int hp = (h == 255) ? 254 : h + 1;
  int wm = (wc == 0)  ? 1   : wc - 1;
  int wp = (wc == 255)? 254 : wc + 1;
  int rm = hm << 8, r0 = h << 8, rp = hp << 8;

  const float* yb = y + ((size_t)n << 22);
  float* ob = out + (n << 22) + q;

#define ADAP_9TAP(B)                                                          \
  (yc[rm + wm] * v[(B)+0] + yc[rm + wc] * v[(B)+1] + yc[rm + wp] * v[(B)+2] + \
   yc[r0 + wm] * v[(B)+3] + yc[r0 + wc] * v[(B)+4] + yc[r0 + wp] * v[(B)+5] + \
   yc[rp + wm] * v[(B)+6] + yc[rp + wc] * v[(B)+7] + yc[rp + wp] * v[(B)+8])

#pragma unroll 4
  for (int c = 0; c < 32; ++c) {       // group 0
    const float* yc = yb + (c << 16);
    ob[c << 16] = ADAP_9TAP(0);
  }
#pragma unroll 4
  for (int c = 32; c < 64; ++c) {      // group 1
    const float* yc = yb + (c << 16);
    ob[c << 16] = ADAP_9TAP(9);
  }
#undef ADAP_9TAP
}

// ---------------------------------------------------------------------------
extern "C" void kernel_launch(void* const* d_in, const int* in_sizes, int n_in,
                              void* d_out, int out_size, void* d_ws, size_t ws_size,
                              hipStream_t stream) {
  const float* y     = (const float*)d_in[0];
  const float* wconv = (const float*)d_in[1];
  const float* gamma = (const float*)d_in[2];
  const float* beta  = (const float*)d_in[3];

  char* ws = (char*)d_ws;
  unsigned short* wfrag = (unsigned short*)ws;           // 36864 B
  float* stats = (float*)(ws + 40960);                   // 36 f32
  float* scsh  = (float*)(ws + 41216);                   // 36 f32
  float* sigma = (float*)(ws + 65536);                   // 4*18*65536 f32 (~18.9 MB)

  float* out  = (float*)d_out;
  float* vmap = out + (size_t)4 * 64 * HW;

  adap_prep<<<72, 256, 0, stream>>>(wconv, wfrag, stats);
  adap_conv_wmma<<<4096, 128, 0, stream>>>(y, wfrag, sigma, stats);
  adap_finalize<<<1, 32, 0, stream>>>(stats, gamma, beta, scsh);
  adap_apply<<<1024, 256, 0, stream>>>(y, sigma, scsh, out, vmap);
}
